// RandomBucketsAttention_34583076667864
// MI455X (gfx1250) — matrix operand order repaired and checked by reading, not compiled, not run on hardware
//
#include <hip/hip_runtime.h>
#include <hip/hip_bf16.h>

typedef _Float16 f16;
typedef __attribute__((ext_vector_type(2)))  __fp16   v2fp16;  // cvt_pkrtz return type
typedef __attribute__((ext_vector_type(16))) _Float16 v16h;
typedef __attribute__((ext_vector_type(8)))  float    v8f;

#define LDS_PITCH 72      // 64 + 8 ushorts; 144B rows: 16B aligned, conflict-free
#define BATCH 8
#define SQLEN 4096
#define SKLEN 4096
#define DHEAD 64
#define NHASH 8

union HFrag { v16h v; uint4 q[2]; };

__device__ __forceinline__ unsigned int pack2h(float a, float b) {
  union { v2fp16 h; unsigned int u; } p;
  p.h = __builtin_amdgcn_cvt_pkrtz(a, b);   // v_cvt_pk_rtz_f16_f32
  return p.u;
}
__device__ __forceinline__ unsigned short h1(float a) {
  union { f16 h; unsigned short u; } p; p.h = (f16)a; return p.u;
}

// A operand (16x32 f16, M x K). Lane L<16: row L, K in {kb..kb+7} U {kb+16..kb+23};
// lanes 16-31: row L-16, K shifted by 8.
__device__ __forceinline__ v16h load_a_frag(const unsigned short* base,
                                            int rowbase, int kb, int lane) {
  const int hlf = lane >> 4;
  const int r   = lane & 15;
  const unsigned short* row = base + (rowbase + r) * LDS_PITCH;
  HFrag f;
  f.q[0] = *(const uint4*)(row + kb + hlf * 8);
  f.q[1] = *(const uint4*)(row + kb + 16 + hlf * 8);
  return f.v;
}

// B operand (32x16 f16, K x N), source stored N-major rows with K contiguous.
// Lane L<16: column L, K in {kb..kb+7} U {kb+16..kb+23}; lanes 16-31: K shifted by 8.
__device__ __forceinline__ v16h load_b_frag(const unsigned short* base,
                                            int colbase, int kb, int lane) {
  const int hlf = lane >> 4;
  const int n   = lane & 15;
  const unsigned short* row = base + (colbase + n) * LDS_PITCH;
  HFrag f;
  f.q[0] = *(const uint4*)(row + kb + hlf * 8);
  f.q[1] = *(const uint4*)(row + kb + 16 + hlf * 8);
  return f.v;
}

__global__ __launch_bounds__(128)
void RandomBucketsAttention_34583076667864_kernel(
    const float* __restrict__ q, const float* __restrict__ k,
    const float* __restrict__ v, const int* __restrict__ kst,
    float* __restrict__ out) {
  __shared__ __align__(16) unsigned short Qs[64 * LDS_PITCH];
  __shared__ __align__(16) unsigned short Ks[64 * LDS_PITCH];
  __shared__ __align__(16) unsigned short Vt[64 * LDS_PITCH];  // transposed: [d][j]
  __shared__ __align__(16) unsigned short Ps[64 * LDS_PITCH];

  const int u    = blockIdx.x >> 6;   // output batch
  const int qc   = blockIdx.x & 63;   // 64-query chunk
  const int tid  = threadIdx.x;
  const int lane = tid & 31;
  const int wv   = tid >> 5;          // wave id: rows [16wv,16wv+16)
  const int hlf  = lane >> 4;

  const int jrow = tid >> 1;          // 0..63: Q row / gathered key row
  const int hcol = tid & 1;           // which 32-float half of the row

  // all-ones B fragment: rowsum(P) = P x ones via WMMA (no LDS, no shuffles)
  v16h ones;
#pragma unroll
  for (int i = 0; i < 16; ++i) ones[i] = (f16)1.0f;

  v8f acc[4] = {};                    // output tiles, f32 accum, dv = 4x16
  float mrow[8], zrow[8], corr[8];
#pragma unroll
  for (int r = 0; r < 8; ++r) { mrow[r] = -1.0e30f; zrow[r] = 0.0f; }

  for (int t = 0; t < NHASH; ++t) {
    // ---- cooperative loads: Q (batch=t!), gathered K rows, gathered V^T ----
    {
      const float4* qsrc =
          (const float4*)(q + (((size_t)t * SQLEN) + qc * 64 + jrow) * DHEAD + hcol * 32);
      unsigned int* qdst = (unsigned int*)(Qs + jrow * LDS_PITCH + hcol * 32);
#pragma unroll
      for (int i = 0; i < 8; ++i) {
        float4 f = qsrc[i];
        qdst[2 * i + 0] = pack2h(f.x, f.y);
        qdst[2 * i + 1] = pack2h(f.z, f.w);
      }
      const int idx =
          kst[(size_t)u * (NHASH * SKLEN) + ((t * 64 + qc) * 64) + jrow];
      const float4* ksrc =
          (const float4*)(k + ((size_t)u * SKLEN + idx) * DHEAD + hcol * 32);
      unsigned int* kdst = (unsigned int*)(Ks + jrow * LDS_PITCH + hcol * 32);
#pragma unroll
      for (int i = 0; i < 8; ++i) {
        float4 f = ksrc[i];
        kdst[2 * i + 0] = pack2h(f.x, f.y);
        kdst[2 * i + 1] = pack2h(f.z, f.w);
      }
      const float4* vsrc =
          (const float4*)(v + ((size_t)u * SKLEN + idx) * DHEAD + hcol * 32);
#pragma unroll
      for (int i = 0; i < 8; ++i) {
        float4 f = vsrc[i];
        const int d = hcol * 32 + i * 4;
        Vt[(d + 0) * LDS_PITCH + jrow] = h1(f.x);
        Vt[(d + 1) * LDS_PITCH + jrow] = h1(f.y);
        Vt[(d + 2) * LDS_PITCH + jrow] = h1(f.z);
        Vt[(d + 3) * LDS_PITCH + jrow] = h1(f.w);
      }
      if (t + 1 < NHASH) {  // global_prefetch_b8 next iteration's gather rows
        const int nidx =
            kst[(size_t)u * (NHASH * SKLEN) + (((t + 1) * 64 + qc) * 64) + jrow];
        __builtin_prefetch(k + ((size_t)u * SKLEN + nidx) * DHEAD + hcol * 32, 0, 0);
        __builtin_prefetch(v + ((size_t)u * SKLEN + nidx) * DHEAD + hcol * 32, 0, 0);
      }
    }
    __syncthreads();

    // ---- S = Q K^T : 4 n-tiles x 2 k-steps of v_wmma_f32_16x16x32_f16 ----
    v16h a0 = load_a_frag(Qs, wv * 16, 0, lane);
    v16h a1 = load_a_frag(Qs, wv * 16, 32, lane);
    v8f st[4];
#pragma unroll
    for (int n = 0; n < 4; ++n) {
      v8f c = {};
      v16h b0 = load_b_frag(Ks, n * 16, 0, lane);
      c = __builtin_amdgcn_wmma_f32_16x16x32_f16(false, a0, false, b0,
                                                 (short)0, c, false, false);
      v16h b1 = load_b_frag(Ks, n * 16, 32, lane);
      c = __builtin_amdgcn_wmma_f32_16x16x32_f16(false, a1, false, b1,
                                                 (short)0, c, false, false);
      st[n] = c;
    }

    // ---- online softmax: row max via 16-lane shuffles, P -> LDS (f16) ----
#pragma unroll
    for (int r = 0; r < 8; ++r) {
      float s0 = st[0][r], s1 = st[1][r], s2 = st[2][r], s3 = st[3][r];
      float mx = fmaxf(fmaxf(s0, s1), fmaxf(s2, s3));
#pragma unroll
      for (int off = 1; off < 16; off <<= 1)
        mx = fmaxf(mx, __shfl_xor(mx, off, 32));
      const float mnew = fmaxf(mrow[r], mx);
      corr[r] = __expf(mrow[r] - mnew);
      mrow[r] = mnew;
      const float p0 = __expf(s0 - mnew), p1 = __expf(s1 - mnew);
      const float p2 = __expf(s2 - mnew), p3 = __expf(s3 - mnew);
      acc[0][r] *= corr[r]; acc[1][r] *= corr[r];
      acc[2][r] *= corr[r]; acc[3][r] *= corr[r];
      unsigned short* prow =
          Ps + (wv * 16 + r + 8 * hlf) * LDS_PITCH + (lane & 15);
      prow[0]  = h1(p0);
      prow[16] = h1(p1);
      prow[32] = h1(p2);
      prow[48] = h1(p3);
    }

    // ---- row sums via WMMA against all-ones B (replaces shuffle reduce) ----
    // (same-wave LDS store->load is in-order; no barrier needed for own rows)
    v16h pa0 = load_a_frag(Ps, wv * 16, 0, lane);
    v16h pa1 = load_a_frag(Ps, wv * 16, 32, lane);
    v8f rsum = {};
    rsum = __builtin_amdgcn_wmma_f32_16x16x32_f16(false, pa0, false, ones,
                                                  (short)0, rsum, false, false);
    rsum = __builtin_amdgcn_wmma_f32_16x16x32_f16(false, pa1, false, ones,
                                                  (short)0, rsum, false, false);
#pragma unroll
    for (int r = 0; r < 8; ++r)
      zrow[r] = zrow[r] * corr[r] + rsum[r];

    // ---- acc += P V ----
#pragma unroll
    for (int n = 0; n < 4; ++n) {
      v16h b0 = load_b_frag(Vt, n * 16, 0, lane);
      acc[n] = __builtin_amdgcn_wmma_f32_16x16x32_f16(false, pa0, false, b0,
                                                      (short)0, acc[n], false, false);
      v16h b1 = load_b_frag(Vt, n * 16, 32, lane);
      acc[n] = __builtin_amdgcn_wmma_f32_16x16x32_f16(false, pa1, false, b1,
                                                      (short)0, acc[n], false, false);
    }
    __syncthreads();  // protect Qs/Ks/Vt/Ps before next iteration's loads
  }

  // ---- out = acc / Z ----
  float* obase = out + ((size_t)u * SQLEN + qc * 64) * DHEAD;
#pragma unroll
  for (int r = 0; r < 8; ++r) {
    const float invz = 1.0f / zrow[r];
    const int row = wv * 16 + r + 8 * hlf;
    float* op = obase + (size_t)row * DHEAD + (lane & 15);
    op[0]  = acc[0][r] * invz;
    op[16] = acc[1][r] * invz;
    op[32] = acc[2][r] * invz;
    op[48] = acc[3][r] * invz;
  }
}

extern "C" void kernel_launch(void* const* d_in, const int* in_sizes, int n_in,
                              void* d_out, int out_size, void* d_ws, size_t ws_size,
                              hipStream_t stream) {
  (void)in_sizes; (void)n_in; (void)out_size; (void)d_ws; (void)ws_size;
  const float* q   = (const float*)d_in[0];
  const float* k   = (const float*)d_in[1];
  const float* v   = (const float*)d_in[2];
  const int*   kst = (const int*)d_in[3];
  float* out = (float*)d_out;

  dim3 grid(BATCH * (SQLEN / 64));  // 512 blocks: (u, query-chunk)
  dim3 block(128);                   // 4 waves of 32
  hipLaunchKernelGGL(RandomBucketsAttention_34583076667864_kernel,
                     grid, block, 0, stream, q, k, v, kst, out);
}
